// Qwen3MoEModel_31636729102465
// MI455X (gfx1250) — matrix operand order repaired and checked by reading, not compiled, not run on hardware
//
#include <hip/hip_runtime.h>
#include <hip/hip_bf16.h>
#include <math.h>

// ---------------------------------------------------------------------------
// Model constants (fixed by the reference)
// ---------------------------------------------------------------------------
#define TT   2048      // B*S tokens
#define HH   1024      // hidden
#define II   1024      // ffn inner
#define SS   1024      // sequence
#define NHq  16        // query heads
#define NKVq 4         // kv heads
#define HDq  64        // head dim
#define EE   8         // experts
#define VV   32000     // vocab

typedef __attribute__((ext_vector_type(16))) _Float16 v16h;
typedef __attribute__((ext_vector_type(8)))  _Float16 v8h;
typedef __attribute__((ext_vector_type(4)))  _Float16 v4h;
typedef __attribute__((ext_vector_type(8)))  float    v8f;

#define EPI_STORE   0
#define EPI_ADD     1
#define EPI_SILUMUL 2

#define BROW 48   // padded LDS row stride (halves); 96B = 0 mod 32 -> v16h aligned

// A fragment: 16x32 f16, lane m=lane&15, half=lane>>4
//   elem e -> K = half*16 + e. A is stored f16, so the two
//   8-half chunks are two 16B global_load_b128 with no conversion.
__device__ __forceinline__ v16h loadA_frag(const _Float16* __restrict__ ap)
{
  union { v16h v; struct { v8h lo, hi; } s; } u;
  u.s.lo = *(const v8h*)(ap);
  u.s.hi = *(const v8h*)(ap + 8);
  return u.v;
}

// ---------------------------------------------------------------------------
// WMMA GEMM:  C(MxN) = A(MxK) * B  (+ epilogue)
//   A: f16 (produced by our own kernels)     B: fp32 weights
//   BT=false: B stored row-major (K,N)   (wq/wk/wv/wo layouts)
//   BT=true : B stored row-major (N,K)   (W.T matmuls: swiglu / lm_head)
// Block = 128 threads (4 waves). Block tile 128x64; wave tile 32x64 (2 A frags
// x 4 B frags = 8 WMMAs per K-step of 32). B tile staged in LDS as f16 (n,k)
// layout, ping-pong double buffered; next tile's global loads issue before the
// current step's WMMAs.
// Epilogues: STORE -> f32 C;  ADD -> f32 C += (scale *) acc;
//            SILUMUL -> f16 Ch = silu(gbuf)*acc  (feeds next GEMM as A)
// ---------------------------------------------------------------------------
template <int EPI, bool BT>
__global__ __launch_bounds__(128) void gemm_wmma(
    const _Float16* __restrict__ A, const float* __restrict__ B,
    float* __restrict__ C, _Float16* __restrict__ Ch, int M, int N, int Kd,
    const float* __restrict__ gbuf,                    // EPI_SILUMUL
    const float* __restrict__ rowscale, int rs_stride) // EPI_ADD
{
  __shared__ __align__(32) _Float16 Bsh[2][64][BROW];

  const int tid   = threadIdx.x;
  const int lane  = tid & 31;
  const int wave  = tid >> 5;
  const int n0    = blockIdx.x * 64;
  const int m0    = blockIdx.y * 128 + wave * 32;   // wave covers 32 rows
  const int mrow  = lane & 15;
  const int lhalf = lane >> 4;

  v8f c[2][4];
#pragma unroll
  for (int mi = 0; mi < 2; ++mi)
#pragma unroll
    for (int i = 0; i < 4; ++i)
      c[mi][i] = (v8f){0.f,0.f,0.f,0.f,0.f,0.f,0.f,0.f};

  const _Float16* Arow0 = A + (size_t)(m0 + mrow) * Kd + lhalf * 16;
  const _Float16* Arow1 = Arow0 + (size_t)16 * Kd;

  // cooperative B-tile fill: 32x64 fp32 -> f16 LDS (n,k) layout
  auto fillB = [&](int k0, int p) {
    if (BT) {
      // B row n is contiguous in k: wide load, wide LDS store
      const int n  = tid >> 1;          // 0..63
      const int ks = (tid & 1) * 16;    // 0 or 16
      const float* brow = B + (size_t)(n0 + n) * Kd + k0 + ks;
      v16h hv;
#pragma unroll
      for (int i = 0; i < 4; ++i) {
        float4 v4 = *(const float4*)(brow + i * 4);
        hv[i*4+0] = (_Float16)v4.x; hv[i*4+1] = (_Float16)v4.y;
        hv[i*4+2] = (_Float16)v4.z; hv[i*4+3] = (_Float16)v4.w;
      }
      *(v16h*)(&Bsh[p][n][ks]) = hv;
    } else {
      // B row k is contiguous in n: wide load, transpose via b16 LDS stores
      const int k  = tid >> 2;          // 0..31
      const int ns = (tid & 3) * 16;    // 0,16,32,48
      const float* brow = B + (size_t)(k0 + k) * N + n0 + ns;
#pragma unroll
      for (int i = 0; i < 4; ++i) {
        float4 v4 = *(const float4*)(brow + i * 4);
        Bsh[p][ns + i*4 + 0][k] = (_Float16)v4.x;
        Bsh[p][ns + i*4 + 1][k] = (_Float16)v4.y;
        Bsh[p][ns + i*4 + 2][k] = (_Float16)v4.z;
        Bsh[p][ns + i*4 + 3][k] = (_Float16)v4.w;
      }
    }
  };

  fillB(0, 0);
  __syncthreads();

  int p = 0;
  for (int k0 = 0; k0 < Kd; k0 += 32) {
    // prefetch next B tile into the other buffer before doing math
    if (k0 + 32 < Kd) fillB(k0 + 32, p ^ 1);

    v16h a0 = loadA_frag(Arow0 + k0);
    v16h a1 = loadA_frag(Arow1 + k0);

    v16h bf[4];
#pragma unroll
    for (int i = 0; i < 4; ++i)
      bf[i] = *(const v16h*)(&Bsh[p][i * 16 + mrow][lhalf * 16]);

#pragma unroll
    for (int i = 0; i < 4; ++i) {
      c[0][i] = __builtin_amdgcn_wmma_f32_16x16x32_f16(
          false, a0, false, bf[i], (short)0, c[0][i], false, false);
      c[1][i] = __builtin_amdgcn_wmma_f32_16x16x32_f16(
          false, a1, false, bf[i], (short)0, c[1][i], false, false);
    }
    __syncthreads();
    p ^= 1;
  }

  // ---- epilogue ----
#pragma unroll
  for (int mi = 0; mi < 2; ++mi) {
    const int mbase = m0 + mi * 16 + lhalf * 8;
#pragma unroll
    for (int i = 0; i < 4; ++i) {
      const int col = n0 + i * 16 + mrow;
#pragma unroll
      for (int r = 0; r < 8; ++r) {
        const size_t idx = (size_t)(mbase + r) * N + col;
        float v = c[mi][i][r];
        if (EPI == EPI_SILUMUL) {
          float g = gbuf[idx];
          Ch[idx] = (_Float16)((g / (1.0f + __expf(-g))) * v);  // silu(g)*u
        } else if (EPI == EPI_ADD) {
          float s = rowscale ? rowscale[(size_t)(mbase + r) * rs_stride] : 1.0f;
          C[idx] += s * v;
        } else {
          C[idx] = v;
        }
      }
    }
  }
}

// ---------------------------------------------------------------------------
// Embedding gather: x[t,:] = embed[ids[t],:]  (float4 copies)
// ---------------------------------------------------------------------------
__global__ __launch_bounds__(256) void embed_k(
    const int* __restrict__ ids, const float* __restrict__ emb,
    float* __restrict__ x)
{
  const int t = blockIdx.x;
  const float4* src = (const float4*)(emb + (size_t)ids[t] * HH);
  float4* dst = (float4*)(x + (size_t)t * HH);
  for (int i = threadIdx.x; i < HH / 4; i += blockDim.x) dst[i] = src[i];
}

// ---------------------------------------------------------------------------
// LayerNorm over H=1024, one block (4 waves) per row; f32 in, f16 out
// (output feeds WMMA GEMMs as the A operand).
// ---------------------------------------------------------------------------
__global__ __launch_bounds__(128) void layernorm_k(
    const float* __restrict__ x, const float* __restrict__ g,
    const float* __restrict__ b, _Float16* __restrict__ y)
{
  __shared__ float rs[4], rs2[4];
  const int row  = blockIdx.x;
  const int lane = threadIdx.x & 31;
  const int wave = threadIdx.x >> 5;
  const float4* xr = (const float4*)(x + (size_t)row * HH);

  float s = 0.f, s2 = 0.f;
  for (int i = threadIdx.x; i < HH / 4; i += blockDim.x) {
    float4 v = xr[i];
    s  += v.x + v.y + v.z + v.w;
    s2 += v.x*v.x + v.y*v.y + v.z*v.z + v.w*v.w;
  }
#pragma unroll
  for (int o = 16; o > 0; o >>= 1) {
    s  += __shfl_xor(s,  o, 32);
    s2 += __shfl_xor(s2, o, 32);
  }
  if (lane == 0) { rs[wave] = s; rs2[wave] = s2; }
  __syncthreads();
  float ts   = rs[0] + rs[1] + rs[2] + rs[3];
  float ts2  = rs2[0] + rs2[1] + rs2[2] + rs2[3];
  float mean = ts * (1.0f / HH);
  float var  = ts2 * (1.0f / HH) - mean * mean;
  float inv  = rsqrtf(var + 1e-5f);

  const float4* gv = (const float4*)g;
  const float4* bv = (const float4*)b;
  _Float16* yr = y + (size_t)row * HH;
  for (int i = threadIdx.x; i < HH / 4; i += blockDim.x) {
    float4 v = xr[i], gg = gv[i], bb = bv[i];
    v4h o4;
    o4[0] = (_Float16)((v.x - mean) * inv * gg.x + bb.x);
    o4[1] = (_Float16)((v.y - mean) * inv * gg.y + bb.y);
    o4[2] = (_Float16)((v.z - mean) * inv * gg.z + bb.z);
    o4[3] = (_Float16)((v.w - mean) * inv * gg.w + bb.w);
    *(v4h*)(yr + i * 4) = o4;
  }
}

// ---------------------------------------------------------------------------
// RoPE in-place on (T, nheads*64): pair (d, d+32), freq = BASE^(-d/32)
// ---------------------------------------------------------------------------
__global__ __launch_bounds__(256) void rope_k(float* __restrict__ x, int nheads)
{
  const int idx = blockIdx.x * blockDim.x + threadIdx.x;
  const int d = idx & 31;
  const int h = (idx >> 5) % nheads;
  const int t = idx / (32 * nheads);
  if (t >= TT) return;
  const int s = t % SS;
  float inv = powf(1000000.0f, -(float)d / 32.0f);
  float ang = (float)s * inv;
  float cs = cosf(ang), sn = sinf(ang);
  float* p = x + (size_t)t * nheads * 64 + h * 64 + d;
  float x1 = p[0], x2 = p[32];
  p[0]  = x1 * cs - x2 * sn;
  p[32] = x2 * cs + x1 * sn;
}

// ---------------------------------------------------------------------------
// Causal GQA attention core. One wave per query row; block = 8 waves.
// Two-pass softmax with scores staged in LDS (8 x 1024 x 4B = 32 KB).
// Output written f16 (feeds the wo GEMM as the A operand).
// ---------------------------------------------------------------------------
#define ATT_ROWS 8
__global__ __launch_bounds__(256) void attn_k(
    const float* __restrict__ Q, const float* __restrict__ Kb,
    const float* __restrict__ Vb, _Float16* __restrict__ O)
{
  __shared__ float sc[ATT_ROWS][SS];
  __shared__ float qsh[ATT_ROWS][64];

  const int bh   = blockIdx.x;
  const int b    = bh / NHq;
  const int h    = bh % NHq;
  const int kvh  = h / (NHq / NKVq);
  const int wave = threadIdx.x >> 5;
  const int lane = threadIdx.x & 31;
  const int s    = blockIdx.y * ATT_ROWS + wave;
  const int t    = b * SS + s;
  const float scale = 0.125f;

  const float* qp = Q + (size_t)t * (NHq * HDq) + h * HDq;
  qsh[wave][lane]      = qp[lane];
  qsh[wave][lane + 32] = qp[lane + 32];
  __syncthreads();

  float m = -1e30f;
  for (int j0 = 0; j0 <= s; j0 += 32) {
    int j = j0 + lane;
    float sv = -1e30f;
    if (j <= s) {
      const float* kp = Kb + (size_t)(b * SS + j) * (NKVq * HDq) + kvh * HDq;
      float acc = 0.f;
#pragma unroll
      for (int d = 0; d < 64; d += 4) {
        float4 k4 = *(const float4*)(kp + d);
        acc += qsh[wave][d]     * k4.x + qsh[wave][d + 1] * k4.y
             + qsh[wave][d + 2] * k4.z + qsh[wave][d + 3] * k4.w;
      }
      sv = acc * scale;
      sc[wave][j] = sv;
    }
    m = fmaxf(m, sv);
  }
#pragma unroll
  for (int o = 16; o > 0; o >>= 1) m = fmaxf(m, __shfl_xor(m, o, 32));

  float l = 0.f;
  for (int j0 = 0; j0 <= s; j0 += 32) {
    int j = j0 + lane;
    if (j <= s) {
      float p = __expf(sc[wave][j] - m);
      sc[wave][j] = p;
      l += p;
    }
  }
#pragma unroll
  for (int o = 16; o > 0; o >>= 1) l += __shfl_xor(l, o, 32);
  const float linv = 1.0f / l;

  const int d0 = lane * 2;
  float o0 = 0.f, o1 = 0.f;
  for (int j = 0; j <= s; ++j) {
    float p = sc[wave][j];
    float2 v2 = *(const float2*)(Vb + (size_t)(b * SS + j) * (NKVq * HDq)
                                 + kvh * HDq + d0);
    o0 += p * v2.x;
    o1 += p * v2.y;
  }
  _Float16* op = O + (size_t)t * (NHq * HDq) + h * HDq + d0;
  op[0] = (_Float16)(o0 * linv);
  op[1] = (_Float16)(o1 * linv);
}

// ---------------------------------------------------------------------------
// MoE gate: logits = x @ gate_w.T (E=8), top-2, softmax over the two.
// X is the f16 LN output.
// ---------------------------------------------------------------------------
__global__ __launch_bounds__(256) void gate_k(
    const _Float16* __restrict__ X, const float* __restrict__ GW,
    float* __restrict__ wexp)
{
  const int wave = threadIdx.x >> 5;
  const int lane = threadIdx.x & 31;
  const int t = blockIdx.x * 8 + wave;
  const _Float16* xr = X + (size_t)t * HH;

  float acc[EE];
#pragma unroll
  for (int e = 0; e < EE; ++e) acc[e] = 0.f;
  for (int i = lane; i < HH; i += 32) {
    float xv = (float)xr[i];
#pragma unroll
    for (int e = 0; e < EE; ++e) acc[e] += xv * GW[e * HH + i];
  }
#pragma unroll
  for (int e = 0; e < EE; ++e)
#pragma unroll
    for (int o = 16; o > 0; o >>= 1) acc[e] += __shfl_xor(acc[e], o, 32);

  if (lane == 0) {
    int e0 = 0; float g0 = acc[0];
    for (int e = 1; e < EE; ++e) if (acc[e] > g0) { g0 = acc[e]; e0 = e; }
    int e1 = -1; float g1 = -1e30f;
    for (int e = 0; e < EE; ++e)
      if (e != e0 && acc[e] > g1) { g1 = acc[e]; e1 = e; }
    float x1 = __expf(g1 - g0);
    float inv = 1.0f / (1.0f + x1);
#pragma unroll
    for (int e = 0; e < EE; ++e) wexp[(size_t)t * EE + e] = 0.f;
    wexp[(size_t)t * EE + e0] = inv;
    wexp[(size_t)t * EE + e1] = x1 * inv;
  }
}

// ---------------------------------------------------------------------------
// Host-side orchestration
// ---------------------------------------------------------------------------
#define FNUL ((float*)nullptr)
#define HNUL ((_Float16*)nullptr)

#define GEMM(EPI, BT, A_, B_, C_, CH_, Mv, Nv, Kv, G_, RS_, RSS_)              \
  gemm_wmma<EPI, BT><<<dim3((Nv) / 64, (Mv) / 128), 128, 0, stream>>>(         \
      A_, B_, C_, CH_, Mv, Nv, Kv, G_, RS_, RSS_)

extern "C" void kernel_launch(void* const* d_in, const int* in_sizes, int n_in,
                              void* d_out, int out_size, void* d_ws, size_t ws_size,
                              hipStream_t stream)
{
  const int*   ids     = (const int*)  d_in[0];
  const float* embed   = (const float*)d_in[1];
  const float* ln1_g   = (const float*)d_in[2];
  const float* ln1_b   = (const float*)d_in[3];
  const float* wq      = (const float*)d_in[4];
  const float* wk      = (const float*)d_in[5];
  const float* wv      = (const float*)d_in[6];
  const float* wo      = (const float*)d_in[7];
  const float* ln2_g   = (const float*)d_in[8];
  const float* ln2_b   = (const float*)d_in[9];
  const float* gate_w  = (const float*)d_in[10];
  const float* exp_wg  = (const float*)d_in[11];
  const float* exp_wu  = (const float*)d_in[12];
  const float* exp_wd  = (const float*)d_in[13];
  const float* sh_wg   = (const float*)d_in[14];
  const float* sh_wu   = (const float*)d_in[15];
  const float* sh_wd   = (const float*)d_in[16];
  const float* fn_g    = (const float*)d_in[17];
  const float* fn_b    = (const float*)d_in[18];
  const float* lm_head = (const float*)d_in[19];
  float* out = (float*)d_out;

  // workspace carve-up
  char* w = (char*)d_ws;
  float* xbuf = (float*)w;      w += (size_t)TT * HH * 4;   // residual (f32)
  float* qbuf = (float*)w;      w += (size_t)TT * HH * 4;   // q (f32, rope'd)
  float* kbuf = (float*)w;      w += (size_t)TT * 256 * 4;  // k (f32, rope'd)
  float* vbuf = (float*)w;      w += (size_t)TT * 256 * 4;  // v (f32)
  float* wexp = (float*)w;      w += (size_t)TT * EE * 4;   // gate weights
  _Float16* hbuf = (_Float16*)w; w += (size_t)TT * HH * 2;  // LN out (f16)
  _Float16* obuf = (_Float16*)w; w += (size_t)TT * HH * 2;  // attn out (f16)
  float* gbuf = qbuf;            // f32 swiglu gate (attn scratch free in MoE)
  _Float16* abuf = obuf;         // f16 swiglu activation (reuse)

  embed_k<<<TT, 256, 0, stream>>>(ids, embed, xbuf);

  for (int l = 0; l < 2; ++l) {
    // ---- attention ----
    layernorm_k<<<TT, 128, 0, stream>>>(xbuf, ln1_g + l * HH, ln1_b + l * HH, hbuf);
    GEMM(EPI_STORE, false, hbuf, wq + (size_t)l * HH * 1024, qbuf, HNUL,
         TT, 1024, HH, FNUL, FNUL, 0);
    GEMM(EPI_STORE, false, hbuf, wk + (size_t)l * HH * 256, kbuf, HNUL,
         TT, 256, HH, FNUL, FNUL, 0);
    GEMM(EPI_STORE, false, hbuf, wv + (size_t)l * HH * 256, vbuf, HNUL,
         TT, 256, HH, FNUL, FNUL, 0);
    rope_k<<<(TT * NHq * 32) / 256, 256, 0, stream>>>(qbuf, NHq);
    rope_k<<<(TT * NKVq * 32) / 256, 256, 0, stream>>>(kbuf, NKVq);
    attn_k<<<dim3(2 * NHq, SS / ATT_ROWS), 256, 0, stream>>>(qbuf, kbuf, vbuf, obuf);
    GEMM(EPI_ADD, false, obuf, wo + (size_t)l * 1024 * HH, xbuf, HNUL,
         TT, HH, 1024, FNUL, FNUL, 0);

    // ---- MoE ----
    layernorm_k<<<TT, 128, 0, stream>>>(xbuf, ln2_g + l * HH, ln2_b + l * HH, hbuf);
    gate_k<<<TT / 8, 256, 0, stream>>>(hbuf, gate_w + (size_t)l * EE * HH, wexp);

    // shared expert: x += swiglu(h)
    GEMM(EPI_STORE,   true, hbuf, sh_wg + (size_t)l * II * HH, gbuf, HNUL,
         TT, II, HH, FNUL, FNUL, 0);
    GEMM(EPI_SILUMUL, true, hbuf, sh_wu + (size_t)l * II * HH, FNUL, abuf,
         TT, II, HH, gbuf, FNUL, 0);
    GEMM(EPI_ADD,     true, abuf, sh_wd + (size_t)l * HH * II, xbuf, HNUL,
         TT, HH, II, FNUL, FNUL, 0);

    // routed experts (dense, weighted by gate): x += w_e * swiglu_e(h)
    for (int e = 0; e < EE; ++e) {
      const float* eg = exp_wg + ((size_t)l * EE + e) * II * HH;
      const float* eu = exp_wu + ((size_t)l * EE + e) * II * HH;
      const float* ed = exp_wd + ((size_t)l * EE + e) * HH * II;
      GEMM(EPI_STORE,   true, hbuf, eg, gbuf, HNUL, TT, II, HH,
           FNUL, FNUL, 0);
      GEMM(EPI_SILUMUL, true, hbuf, eu, FNUL, abuf, TT, II, HH,
           gbuf, FNUL, 0);
      GEMM(EPI_ADD,     true, abuf, ed, xbuf, HNUL, TT, HH, II,
           FNUL, wexp + e, EE);
    }
  }

  // ---- final norm + lm_head ----
  layernorm_k<<<TT, 128, 0, stream>>>(xbuf, fn_g, fn_b, hbuf);
  GEMM(EPI_STORE, true, hbuf, lm_head, out, HNUL, TT, VV, HH,
       FNUL, FNUL, 0);
}
